// BERTFactorizedAttention_35201551958696
// MI455X (gfx1250) — matrix operand order repaired and checked by reading, not compiled, not run on hardware
//
#include <hip/hip_runtime.h>

#define B_  8
#define S_  2048
#define H_  1024
#define NH_ 16
#define DH_ 64
#define M_  (B_*S_)   // 16384 rows for the QKV GEMMs

typedef __attribute__((ext_vector_type(16))) __bf16       v16bf;
typedef __attribute__((ext_vector_type(8)))  float        v8f;
typedef __attribute__((ext_vector_type(4)))  unsigned int v4u;
typedef __attribute__((ext_vector_type(8)))  unsigned int v8u;

// f32 -> bf16 round-to-nearest-even
__device__ __forceinline__ unsigned short f2bf(float f) {
  unsigned int u = __builtin_bit_cast(unsigned int, f);
  u += 0x7fffu + ((u >> 16) & 1u);
  return (unsigned short)(u >> 16);
}

// Assemble a 16xbf16 fragment from two 16B-aligned LDS runs of 8 bf16 each.
__device__ __forceinline__ v16bf frag_from(const unsigned short* p0,
                                           const unsigned short* p1) {
  v4u x = *(const v4u*)p0;
  v4u y = *(const v4u*)p1;
  v8u u;
  u[0] = x[0]; u[1] = x[1]; u[2] = x[2]; u[3] = x[3];
  u[4] = y[0]; u[5] = y[1]; u[6] = y[2]; u[7] = y[3];
  return __builtin_bit_cast(v16bf, u);
}

// LDS byte offset of a __shared__ object: flat aperture maps LDS at addr[31:0].
__device__ __forceinline__ unsigned lds_off(const void* p) {
  return (unsigned)(uintptr_t)p;
}

// Async global->LDS copy of 16 bytes per lane (ASYNCcnt-tracked, no VGPR data).
__device__ __forceinline__ void async_copy_b128(unsigned lds_byte,
                                                const float* gptr) {
  asm volatile("global_load_async_to_lds_b128 %0, %1, off"
               :: "v"(lds_byte), "v"((unsigned long long)(uintptr_t)gptr)
               : "memory");
}

// ---------------------------------------------------------------------------
// Kernel 1: QKV projection GEMM.  Out[M,H] = X[M,H] @ W[H,H] + bias
// 128x128x32 block tiles, 8 waves, each wave computes 64x32 via bf16 WMMA.
// ---------------------------------------------------------------------------
#define TM 128
#define TN 128
#define TK 32
#define LDT 40   // LDS pitch in bf16 (80B, 16B-aligned, conflict-skewed)

__global__ __launch_bounds__(256) void qkv_gemm(const float* __restrict__ X,
                                                const float* __restrict__ W,
                                                const float* __restrict__ bias,
                                                float* __restrict__ Out) {
  __shared__ __attribute__((aligned(16))) unsigned short sA[TM * LDT]; // [m][k]
  __shared__ __attribute__((aligned(16))) unsigned short sB[TN * LDT]; // [n][k] (W transposed)
  const int tid  = threadIdx.x;
  const int wid  = tid >> 5;
  const int lane = tid & 31;
  const int l16  = lane & 15;
  const int g    = lane >> 4;          // K-group of this lane
  const int m0   = blockIdx.y * TM;
  const int n0   = blockIdx.x * TN;
  const int wm   = (wid & 1) * 64;     // wave tile origin inside block
  const int wn   = (wid >> 1) * 32;

  v8f acc[4][2];
  for (int i = 0; i < 4; i++)
    for (int j = 0; j < 2; j++)
      for (int r = 0; r < 8; r++) acc[i][j][r] = 0.f;

  for (int kt = 0; kt < H_; kt += TK) {
    // Stage A tile (f32 -> bf16)
    for (int idx = tid; idx < TM * TK; idx += 256) {
      int r = idx >> 5, c = idx & 31;
      sA[r * LDT + c] = f2bf(X[(size_t)(m0 + r) * H_ + kt + c]);
    }
    // Stage B tile transposed: sB[n][k] from W[k][n]
    for (int idx = tid; idx < TK * TN; idx += 256) {
      int kr = idx >> 7, n = idx & 127;
      sB[n * LDT + kr] = f2bf(W[(size_t)(kt + kr) * H_ + n0 + n]);
    }
    __syncthreads();

    v16bf afrag[4], bfrag[2];
    for (int i = 0; i < 4; i++) {
      const unsigned short* base = &sA[(wm + i * 16 + l16) * LDT];
      afrag[i] = frag_from(base + g * 8, base + 16 + g * 8);   // K = g*8+i, 16+g*8+i
    }
    for (int j = 0; j < 2; j++) {
      const unsigned short* base = &sB[(wn + j * 16 + l16) * LDT];
      bfrag[j] = frag_from(base + g * 8, base + 16 + g * 8);   // K = g*8+i, 16+g*8+i
    }
    for (int i = 0; i < 4; i++)
      for (int j = 0; j < 2; j++)
        acc[i][j] = __builtin_amdgcn_wmma_f32_16x16x32_bf16(
            false, afrag[i], false, bfrag[j], (short)0, acc[i][j], false, false);
    __syncthreads();
  }

  // D layout: m = r + 8*g, n = l16
  for (int j = 0; j < 2; j++) {
    int n = n0 + wn + j * 16 + l16;
    float bb = bias[n];
    for (int i = 0; i < 4; i++) {
      int mbase = m0 + wm + i * 16 + g * 8;
      for (int r = 0; r < 8; r++)
        Out[(size_t)(mbase + r) * H_ + n] = acc[i][j][r] + bb;
    }
  }
}

// ---------------------------------------------------------------------------
// Kernel 2: softmax over sequence for q viewed as [B,NH,DH,S] (in place).
// One block per (b,head); 64 lanes span DH (contiguous!), 4 groups span S.
// ---------------------------------------------------------------------------
__global__ __launch_bounds__(256) void softmax_q(float* __restrict__ q,
                                                 const float* __restrict__ mask) {
  __shared__ float smx[256], ssum[256];
  const int bx = blockIdx.x;
  const int b = bx / NH_, h = bx % NH_;
  float* base = q + (size_t)b * S_ * H_ + h * DH_;
  const float* mrow = mask + (size_t)b * S_;
  const int d  = threadIdx.x & 63;
  const int sg = threadIdx.x >> 6;

  float mx = -3.0e38f, sum = 0.f;
  for (int s = sg; s < S_; s += 4) {
    float v = base[(size_t)s * H_ + d] + mrow[s];
    float nm = fmaxf(mx, v);
    sum = sum * __expf(mx - nm) + __expf(v - nm);
    mx = nm;
  }
  smx[threadIdx.x] = mx; ssum[threadIdx.x] = sum;
  __syncthreads();
  if (sg == 0) {
    for (int gg = 1; gg < 4; gg++) {
      float m2 = smx[gg * 64 + d], s2 = ssum[gg * 64 + d];
      float nm = fmaxf(mx, m2);
      sum = sum * __expf(mx - nm) + s2 * __expf(m2 - nm);
      mx = nm;
    }
    smx[d] = mx; ssum[d] = sum;
  }
  __syncthreads();
  mx = smx[d];
  float inv = 1.0f / ssum[d];
  for (int s = sg; s < S_; s += 4) {
    size_t off = (size_t)s * H_ + d;
    base[off] = __expf(base[off] + mrow[s] - mx) * inv;
  }
}

// ---------------------------------------------------------------------------
// Kernel 3: softmax over head-dim (64 contiguous floats) — one wave32 per row.
// ---------------------------------------------------------------------------
__global__ __launch_bounds__(256) void softmax_k(float* __restrict__ k) {
  const int row  = blockIdx.x * 8 + (threadIdx.x >> 5);
  const int lane = threadIdx.x & 31;
  float* base = k + (size_t)row * 64;
  float v0 = base[lane], v1 = base[lane + 32];
  float m = fmaxf(v0, v1);
  for (int off = 16; off > 0; off >>= 1) m = fmaxf(m, __shfl_xor(m, off, 32));
  float e0 = __expf(v0 - m), e1 = __expf(v1 - m);
  float s = e0 + e1;
  for (int off = 16; off > 0; off >>= 1) s += __shfl_xor(s, off, 32);
  float inv = 1.0f / s;
  base[lane]      = e0 * inv;
  base[lane + 32] = e1 * inv;
}

// ---------------------------------------------------------------------------
// Kernel 4: s_ctx[b,h,d,e] = sum_s q[b,h,d,s] * v[b,h,s,e]   (64x64, K=2048)
// Memory-bound (reads 128 MB). f32 tiles staged with double-buffered
// GLOBAL_LOAD_ASYNC_TO_LDS_B128 (ASYNCcnt) to overlap HBM with the VALU
// outer-product accumulation.
// ---------------------------------------------------------------------------
__global__ __launch_bounds__(256) void sctx_kernel(const float* __restrict__ q,
                                                   const float* __restrict__ v,
                                                   float* __restrict__ sctx) {
  __shared__ __attribute__((aligned(16))) float Qs[2][32 * 64];
  __shared__ __attribute__((aligned(16))) float Vs[2][32 * 64];
  const int bh = blockIdx.x;
  const int b = bh / NH_, h = bh % NH_;
  const float* qb = q + (size_t)b * S_ * H_ + h * DH_;
  const float* vb = v + (size_t)b * S_ * H_ + h * DH_;
  const int tid = threadIdx.x;
  const int d0 = (tid >> 4) * 4, e0 = (tid & 15) * 4;

  float acc[4][4];
  for (int i = 0; i < 4; i++) for (int j = 0; j < 4; j++) acc[i][j] = 0.f;

  const unsigned qs0 = lds_off(&Qs[0][0]);
  const unsigned vs0 = lds_off(&Vs[0][0]);

  // Issue one 32-row chunk (8KB Q + 8KB V) as 4 b128 async copies per thread.
  auto issue = [&](int sc, int buf) {
    for (int t = 0; t < 2; ++t) {
      int id = tid + t * 256;          // 0..511 float4 slots
      int r  = id >> 4;                // row within chunk
      int c4 = id & 15;                // float4 column
      unsigned loff = (unsigned)(buf * 32 * 64 * 4 + (r * 64 + c4 * 4) * 4);
      const float* gq = qb + (size_t)(sc + r) * H_ + c4 * 4;
      const float* gv = vb + (size_t)(sc + r) * H_ + c4 * 4;
      async_copy_b128(qs0 + loff, gq);
      async_copy_b128(vs0 + loff, gv);
    }
  };

  issue(0, 0);
  const int nChunks = S_ / 32;   // 64
  for (int c = 0; c < nChunks; ++c) {
    const int cur = c & 1;
    if (c + 1 < nChunks) {
      issue((c + 1) * 32, cur ^ 1);
      // Wait for the current chunk only; leave the 4 prefetch copies in flight.
      asm volatile("s_wait_asynccnt 0x4" ::: "memory");
    } else {
      asm volatile("s_wait_asynccnt 0x0" ::: "memory");
    }
    __syncthreads();
    const float* Qc = &Qs[cur][0];
    const float* Vc = &Vs[cur][0];
    for (int ss = 0; ss < 32; ss++) {
      float qv[4], vv[4];
      for (int i = 0; i < 4; i++) qv[i] = Qc[ss * 64 + d0 + i];
      for (int j = 0; j < 4; j++) vv[j] = Vc[ss * 64 + e0 + j];
      for (int i = 0; i < 4; i++)
        for (int j = 0; j < 4; j++) acc[i][j] += qv[i] * vv[j];
    }
    __syncthreads();
  }
  float* ob = sctx + (size_t)bh * 64 * 64;
  for (int i = 0; i < 4; i++)
    for (int j = 0; j < 4; j++)
      ob[(d0 + i) * 64 + e0 + j] = acc[i][j];
}

// ---------------------------------------------------------------------------
// Kernel 5: ctx[b,s,h*64+e] = sum_d c_probs[b,s,h*64+d] * s_ctx[b,h,d,e]
// Per (b,h, 128-row s-block): stage c_probs + s_ctx^T as bf16, WMMA M=16 N=64 K=64.
// ---------------------------------------------------------------------------
#define LDC 72   // pitch in bf16 (144B, 16B-aligned)
__global__ __launch_bounds__(256) void ctx_wmma(const float* __restrict__ cp,
                                                const float* __restrict__ sctx,
                                                float* __restrict__ out) {
  __shared__ __attribute__((aligned(16))) unsigned short sA[128 * LDC]; // [s][d]
  __shared__ __attribute__((aligned(16))) unsigned short sB[64 * LDC];  // [e][d]
  const int bh = blockIdx.y;
  const int b = bh / NH_, h = bh % NH_;
  const int s0 = blockIdx.x * 128;
  const int tid  = threadIdx.x;
  const int wid  = tid >> 5;
  const int lane = tid & 31;
  const int l16  = lane & 15, g = lane >> 4;
  const float* cb = cp + (size_t)b * S_ * H_ + h * DH_;
  const float* sb = sctx + (size_t)bh * 64 * 64;

  for (int idx = tid; idx < 128 * 64; idx += 256) {
    int r = idx >> 6, c = idx & 63;
    sA[r * LDC + c] = f2bf(cb[(size_t)(s0 + r) * H_ + c]);
  }
  for (int idx = tid; idx < 64 * 64; idx += 256) {
    int r = idx >> 6, c = idx & 63;            // r = d, c = e
    sB[c * LDC + r] = f2bf(sb[idx]);           // store transposed [e][d]
  }
  __syncthreads();

  v8f acc[4];
  for (int j = 0; j < 4; j++)
    for (int r = 0; r < 8; r++) acc[j][r] = 0.f;

  for (int k0 = 0; k0 < 64; k0 += 32) {
    const unsigned short* abase = &sA[(wid * 16 + l16) * LDC + k0];
    v16bf afrag = frag_from(abase + g * 8, abase + 16 + g * 8);
    for (int j = 0; j < 4; j++) {
      const unsigned short* bbase = &sB[(j * 16 + l16) * LDC + k0];
      v16bf bfrag = frag_from(bbase + g * 8, bbase + 16 + g * 8);
      acc[j] = __builtin_amdgcn_wmma_f32_16x16x32_bf16(
          false, afrag, false, bfrag, (short)0, acc[j], false, false);
    }
  }

  float* ob = out + (size_t)b * S_ * H_ + h * DH_;
  for (int j = 0; j < 4; j++) {
    int e = j * 16 + l16;
    int mb = wid * 16 + g * 8;
    for (int r = 0; r < 8; r++)
      ob[(size_t)(s0 + mb + r) * H_ + e] = acc[j][r];
  }
}

// ---------------------------------------------------------------------------
extern "C" void kernel_launch(void* const* d_in, const int* in_sizes, int n_in,
                              void* d_out, int out_size, void* d_ws, size_t ws_size,
                              hipStream_t stream) {
  const float* hs   = (const float*)d_in[0];
  const float* mask = (const float*)d_in[1];
  const float* Wq   = (const float*)d_in[2];
  const float* bq   = (const float*)d_in[3];
  const float* Wk   = (const float*)d_in[4];
  const float* bk   = (const float*)d_in[5];
  const float* Wv   = (const float*)d_in[6];
  const float* bv   = (const float*)d_in[7];
  float* out = (float*)d_out;

  size_t nElem = (size_t)M_ * H_;          // 16M floats per buffer
  float* qbuf = (float*)d_ws;
  float* kbuf = qbuf + nElem;
  float* vbuf = kbuf + nElem;
  float* sctx = vbuf + nElem;              // 128*64*64 floats

  dim3 gGemm(H_ / TN, M_ / TM);            // (8, 128)
  qkv_gemm<<<gGemm, 256, 0, stream>>>(hs, Wq, bq, qbuf);
  qkv_gemm<<<gGemm, 256, 0, stream>>>(hs, Wk, bk, kbuf);
  qkv_gemm<<<gGemm, 256, 0, stream>>>(hs, Wv, bv, vbuf);

  softmax_q<<<B_ * NH_, 256, 0, stream>>>(qbuf, mask);
  softmax_k<<<(B_ * S_ * NH_) / 8, 256, 0, stream>>>(kbuf);

  sctx_kernel<<<B_ * NH_, 256, 0, stream>>>(qbuf, vbuf, sctx);
  ctx_wmma<<<dim3(S_ / 128, B_ * NH_), 256, 0, stream>>>(kbuf, sctx, out);
}
